// STAE_80118319939738
// MI455X (gfx1250) — compile-verified, matrix operand order checked by reading, not run on hardware
//
#include <hip/hip_runtime.h>
#include <hip/hip_bf16.h>

typedef __bf16 bf16_t;
typedef __attribute__((ext_vector_type(16))) __bf16 v16bf;
typedef __attribute__((ext_vector_type(8)))  __bf16 v8bf;
typedef __attribute__((ext_vector_type(8)))  float  v8f;

#define HW   9216
#define WID  96
#define PIX_STRIDE 40    // LDS channel-minor pixel stride (80B: 16B-aligned, bank-spread)
#define BW_STRIDE  296   // LDS weight row stride (592B: 16B-aligned, bank-spread)

// ---------------- pack x = [flow, 0.5*down2(img_k), 0.5*res] as bf16 ----------------
__global__ __launch_bounds__(256)
void pack_x_kernel(const float* __restrict__ flow,
                   const float* __restrict__ img,
                   const float* __restrict__ res,
                   bf16_t* __restrict__ X) {
  size_t idx = (size_t)blockIdx.x * 256 + threadIdx.x;
  size_t total = (size_t)2 * 4 * 768 * HW;
  if (idx >= total) return;
  int hw = (int)(idx % HW);
  int c  = (int)((idx / HW) % 768);
  int n  = (int)((idx / ((size_t)HW * 768)) % 4);
  int v  = (int)(idx / ((size_t)HW * 768 * 4));
  float val;
  if (c < 256) {
    val = flow[((size_t)n * 256 + c) * HW + hw];
  } else if (c < 512) {
    int ci = c - 256;
    int h = hw / WID, w = hw % WID;
    const float* ip = img + (((size_t)(2 * n + v) * 256 + ci) * 192 + 2 * h) * 192 + 2 * w;
    // half-pixel bilinear 2x downsample == 2x2 mean; fold in the 0.5 concat scale
    val = 0.125f * (ip[0] + ip[1] + ip[192] + ip[193]);
  } else {
    val = 0.5f * res[((size_t)n * 256 + (c - 512)) * HW + hw];
  }
  X[idx] = (bf16_t)val;
}

// ---------------- repack OIHW f32 weights -> bf16 [CoutPad][tap][CinPad] ----------------
__global__ __launch_bounds__(256)
void pack_w_kernel(const float* __restrict__ src, bf16_t* __restrict__ dst,
                   int Cout, int CoutPad, int Cin, int CinPad) {
  size_t idx = (size_t)blockIdx.x * 256 + threadIdx.x;
  size_t total = (size_t)CoutPad * 9 * CinPad;
  if (idx >= total) return;
  int c   = (int)(idx % CinPad);
  int tap = (int)((idx / CinPad) % 9);
  int o   = (int)(idx / ((size_t)CinPad * 9));
  float v = 0.f;
  if (o < Cout && c < Cin) v = src[((size_t)o * Cin + c) * 9 + tap];
  dst[idx] = (bf16_t)v;
}

__global__ void zero_kernel(float* __restrict__ p, int n) {
  int i = blockIdx.x * 256 + threadIdx.x;
  if (i < n) p[i] = 0.f;
}

// ---------------- implicit-GEMM 3x3 SAME conv via WMMA bf16 ----------------
// block: 192 threads = 6 wave32 waves; tile: M=96 (one image row), N = NT*16 out-chans.
// Per 32-channel block: stage [3 rows][98 px][32 ch] once, run all 9 taps x NT WMMAs.
// Double-buffered LDS -> one barrier per block; staging overlaps tensor-core work.
// mode 0: bias + leakyReLU(0.1), append growth channels into G.
// mode 1: fused global average pool -> atomicAdd into pooled[v][n][c].
// mode 2: write column 0 (+bias) to spatial map.
template <int NT>
__global__ __launch_bounds__(192)
void conv_wmma_kernel(const bf16_t* __restrict__ X,
                      bf16_t* __restrict__ G,           // growth buf for this predictor: [v][n][28][HW]
                      const bf16_t* __restrict__ Wp,    // [CoutPad][9][CinPad]
                      const float*  __restrict__ bias,
                      int Cin, int CinPad, int gIn, int gOff,
                      int mode, int growth, int gOutBase,
                      float* __restrict__ pooled,
                      float* __restrict__ swraw) {
  __shared__ __align__(16) bf16_t Ax[2][3 * 98 * PIX_STRIDE];
  __shared__ __align__(16) bf16_t Bw[2][16 * NT * BW_STRIDE];

  const int t  = threadIdx.x;
  const int v  = blockIdx.z;
  const int n  = blockIdx.x / WID;
  const int h  = blockIdx.x % WID;
  const int n0 = blockIdx.y * (16 * NT);
  const int Kpad = 9 * CinPad;

  const bf16_t* Xv = X + (size_t)(v * 4 + n) * 768 * HW;
  bf16_t*       Gv = G + (size_t)(v * 4 + n) * 28 * HW;

  const int lane = t & 31;
  const int wv   = t >> 5;                 // 0..5 (wave id)
  const int kb   = (lane >> 4) << 3;       // 0 or 8 (lane-half K base)
  const int mrow = wv * 16 + (lane & 15);  // output pixel this lane owns (A rows)

  // ---- stage one 32-channel input block: [3 rows][98 px][32 ch] ----
  auto stageA = [&](int cb, bf16_t* A) {
    const int u    = t >> 1;     // 0..95 : (row, channel) unit
    const int half = t & 1;      // 49 pixels each
    const int r    = u >> 5;     // 0..2
    const int kk   = u & 31;     // channel slot
    const int c    = cb + kk;
    const int hin  = h + r - 1;
    const bf16_t* src = nullptr;
    if (c < Cin && (unsigned)hin < 96u)
      src = (c < gIn) ? (Gv + (size_t)(gOff + c) * HW + hin * WID)
                      : (Xv + (size_t)(c - gIn) * HW + hin * WID);
    bf16_t* dst = A + (r * 98) * PIX_STRIDE + kk;
    // edge pixel (w = -1 or w = 96) is always zero (SAME padding)
    dst[(half ? 97 : 0) * PIX_STRIDE] = (bf16_t)0.f;
    const int pbase = half ? 49 : 1;         // 48 interior pixels each
    const int wbase = pbase - 1;
    #pragma unroll
    for (int ch = 0; ch < 6; ++ch) {
      v8bf vec = {(bf16_t)0.f, (bf16_t)0.f, (bf16_t)0.f, (bf16_t)0.f,
                  (bf16_t)0.f, (bf16_t)0.f, (bf16_t)0.f, (bf16_t)0.f};
      if (src) vec = *(const v8bf*)(src + wbase + ch * 8);   // 16B global load
      #pragma unroll
      for (int e = 0; e < 8; ++e)
        dst[(pbase + ch * 8 + e) * PIX_STRIDE] = vec[e];
    }
  };
  // ---- stage weights for this channel block, all 9 taps: [16*NT oc][9 tap][32 ch] ----
  auto stageB = [&](int cb, bf16_t* B) {
    for (int q = t; q < 16 * NT * 36; q += 192) {    // 16B vectors
      const int j   = q / 36;
      const int rem = q - j * 36;
      const int tap = rem >> 2;
      const int ch4 = rem & 3;
      const uint4* s = (const uint4*)(Wp + (size_t)(n0 + j) * Kpad + tap * CinPad + cb + ch4 * 8);
      *((uint4*)&B[j * BW_STRIDE + tap * 32 + ch4 * 8]) = *s;
    }
  };

  v8f acc[NT];
  #pragma unroll
  for (int nt = 0; nt < NT; ++nt)
    acc[nt] = (v8f){0.f, 0.f, 0.f, 0.f, 0.f, 0.f, 0.f, 0.f};

  const int nblk = CinPad / 32;
  stageA(0, Ax[0]);
  stageB(0, Bw[0]);

  for (int blk = 0; blk < nblk; ++blk) {
    __syncthreads();                               // staged block visible to all
    if (blk + 1 < nblk) {                          // overlap next staging with WMMAs
      stageA((blk + 1) * 32, Ax[(blk + 1) & 1]);
      stageB((blk + 1) * 32, Bw[(blk + 1) & 1]);
    }
    if (t == 0 && blk + 2 < nblk)
      __builtin_prefetch(Wp + (size_t)n0 * Kpad + (blk + 2) * 32, 0, 1);

    const bf16_t* A = Ax[blk & 1];
    const bf16_t* B = Bw[blk & 1];
    #pragma unroll
    for (int tap = 0; tap < 9; ++tap) {
      const int dr = tap / 3;
      const int dc = tap % 3;
      // A fragment: lane row = output pixel mrow, shifted window (pixel idx = w+1)
      const bf16_t* ap = &A[((dr * 98) + mrow + dc) * PIX_STRIDE + kb];
      v8bf a0 = *(const v8bf*)ap;
      v8bf a1 = *(const v8bf*)(ap + 16);
      v16bf a = __builtin_shufflevector(a0, a1, 0,1,2,3,4,5,6,7,8,9,10,11,12,13,14,15);
      #pragma unroll
      for (int nt = 0; nt < NT; ++nt) {            // A fragment reused NT times
        const bf16_t* bp = &B[(nt * 16 + (lane & 15)) * BW_STRIDE + tap * 32 + kb];
        v8bf b0 = *(const v8bf*)bp;
        v8bf b1 = *(const v8bf*)(bp + 16);
        v16bf b = __builtin_shufflevector(b0, b1, 0,1,2,3,4,5,6,7,8,9,10,11,12,13,14,15);
        acc[nt] = __builtin_amdgcn_wmma_f32_16x16x32_bf16(false, a, false, b, (short)0,
                                                          acc[nt], false, false);
      }
    }
  }

  // ---- epilogue; C/D layout: elem e -> M = e + 8*(lane>=16), N = lane&15 ----
  const int col = lane & 15;
  const int hi  = lane >> 4;
  const int m0  = wv * 16;

  if (mode == 0) {
    if (col < growth) {                            // NT==1 path
      const float bv = bias[col];
      bf16_t* dst = Gv + (size_t)(gOutBase + col) * HW + h * WID;
      #pragma unroll
      for (int e = 0; e < 8; ++e) {
        const int w = m0 + e + hi * 8;
        const float x = acc[0][e] + bv;
        dst[w] = (bf16_t)(x > 0.f ? x : 0.1f * x);
      }
    }
  } else if (mode == 1) {
    #pragma unroll
    for (int nt = 0; nt < NT; ++nt) {
      float s = 0.f;
      #pragma unroll
      for (int e = 0; e < 8; ++e) s += acc[nt][e];
      s += __shfl_xor(s, 16, 32);          // fold the two lane-halves (rows 0-7 + 8-15)
      if (lane < 16)
        atomicAdd(&pooled[(size_t)(v * 4 + n) * 256 + n0 + nt * 16 + col], s);
    }
  } else {
    if (col == 0) {                                // NT==1 path
      const float bv = bias[0];
      #pragma unroll
      for (int e = 0; e < 8; ++e) {
        const int w = m0 + e + hi * 8;
        swraw[(size_t)(v * 4 + n) * HW + h * WID + w] = acc[0][e] + bv;
      }
    }
  }
}

// ---------------- MLP: wgt = sigmoid(relu(pm @ W1^T + b1) @ W2^T + b2) ----------------
__global__ __launch_bounds__(256)
void mlp_kernel(const float* __restrict__ pooled, const float* __restrict__ cb5,
                const float* __restrict__ W1, const float* __restrict__ b1,
                const float* __restrict__ W2, const float* __restrict__ b2,
                float* __restrict__ wgt) {
  __shared__ float pm[4 * 256];
  __shared__ float hs[4 * 256];
  const int v = blockIdx.x;
  const int o = threadIdx.x;
  for (int n = 0; n < 4; ++n)
    pm[n * 256 + o] = pooled[(size_t)(v * 4 + n) * 256 + o] * (1.f / 9216.f) + cb5[o];
  __syncthreads();
  for (int n = 0; n < 4; ++n) {
    float s = b1[o];
    for (int i = 0; i < 256; ++i) s += pm[n * 256 + i] * W1[o * 256 + i];
    hs[n * 256 + o] = s > 0.f ? s : 0.f;
  }
  __syncthreads();
  for (int n = 0; n < 4; ++n) {
    float s = b2[o];
    for (int i = 0; i < 256; ++i) s += hs[n * 256 + i] * W2[o * 256 + i];
    wgt[(size_t)(v * 4 + n) * 256 + o] = 1.f / (1.f + __expf(-s));
  }
}

// ---------------- spatial softmax over 9216, in place ----------------
__global__ __launch_bounds__(256)
void softmax_kernel(float* __restrict__ sw) {
  __shared__ float red[256];
  float* p = sw + (size_t)blockIdx.x * HW;
  const int t = threadIdx.x;
  float mx = -3.4e38f;
  for (int i = t; i < HW; i += 256) mx = fmaxf(mx, p[i]);
  red[t] = mx; __syncthreads();
  for (int s = 128; s > 0; s >>= 1) { if (t < s) red[t] = fmaxf(red[t], red[t + s]); __syncthreads(); }
  mx = red[0]; __syncthreads();
  float sum = 0.f;
  for (int i = t; i < HW; i += 256) sum += __expf(p[i] - mx);
  red[t] = sum; __syncthreads();
  for (int s = 128; s > 0; s >>= 1) { if (t < s) red[t] += red[t + s]; __syncthreads(); }
  const float inv = 1.f / red[0];
  for (int i = t; i < HW; i += 256) p[i] = __expf(p[i] - mx) * inv;
}

// ---------------- out = flow + 0.5*res*(w0*s0 + w1*s1) ----------------
__global__ __launch_bounds__(256)
void combine_kernel(const float* __restrict__ flow, const float* __restrict__ res,
                    const float* __restrict__ wgt, const float* __restrict__ sw,
                    float* __restrict__ out) {
  size_t idx = (size_t)blockIdx.x * 256 + threadIdx.x;
  const size_t total = (size_t)4 * 256 * HW;
  if (idx >= total) return;
  const int hw = (int)(idx % HW);
  const int c  = (int)((idx / HW) % 256);
  const int n  = (int)(idx / ((size_t)HW * 256));
  const float s0 = sw[(size_t)n * HW + hw];
  const float s1 = sw[(size_t)(4 + n) * HW + hw];
  const float w0 = wgt[(size_t)n * 256 + c];
  const float w1 = wgt[(size_t)(4 + n) * 256 + c];
  out[idx] = flow[idx] + 0.5f * res[idx] * (w0 * s0 + w1 * s1);
}

extern "C" void kernel_launch(void* const* d_in, const int* in_sizes, int n_in,
                              void* d_out, int out_size, void* d_ws, size_t ws_size,
                              hipStream_t stream) {
  const float* flow = (const float*)d_in[0];
  const float* img  = (const float*)d_in[1];
  const float* res  = (const float*)d_in[2];
  const float *chW[6], *chB[6], *spW[6], *spB[6];
  for (int i = 0; i < 6; ++i) {
    chW[i] = (const float*)d_in[3 + i];
    chB[i] = (const float*)d_in[9 + i];
    spW[i] = (const float*)d_in[15 + i];
    spB[i] = (const float*)d_in[21 + i];
  }
  const float* W1 = (const float*)d_in[27];
  const float* b1 = (const float*)d_in[28];
  const float* W2 = (const float*)d_in[29];
  const float* b2 = (const float*)d_in[30];

  // ---- carve workspace ----
  char* ws = (char*)d_ws;
  size_t off = 0;
  auto carve = [&](size_t bytes) -> char* {
    off = (off + 255) & ~(size_t)255;
    char* p = ws + off;
    off += bytes;
    return p;
  };
  bf16_t* X = (bf16_t*)carve((size_t)2 * 4 * 768 * HW * 2);      // packed x, both variants
  bf16_t* G = (bf16_t*)carve((size_t)2 * 2 * 4 * 28 * HW * 2);   // [pred][v][n][28][HW]

  // channel layout of the 28 growth channels: [y4(0-1) y3(2-5) y2(6-11) y1(12-19) y0(20-27)]
  struct CFG { int Cin, CinPad, gIn, gOff, gOutBase; };
  const CFG cfg[6] = {
    {768, 768,  0,  0, 20},
    {776, 800,  8, 20, 12},
    {784, 800, 16, 12,  6},
    {790, 800, 22,  6,  2},
    {794, 800, 26,  2,  0},
    {796, 800, 28,  0,  0},
  };
  const int grow[5] = {8, 8, 6, 4, 2};

  bf16_t *WpCh[6], *WpSp[6];
  for (int k = 0; k < 6; ++k) {
    const int coCp = (k < 5) ? 16 : 256;
    WpCh[k] = (bf16_t*)carve((size_t)coCp * 9 * cfg[k].CinPad * 2);
    WpSp[k] = (bf16_t*)carve((size_t)16 * 9 * cfg[k].CinPad * 2);
  }
  float* pooled = (float*)carve((size_t)2 * 4 * 256 * 4);
  float* wgt    = (float*)carve((size_t)2 * 4 * 256 * 4);
  float* sw     = (float*)carve((size_t)2 * 4 * HW * 4);

  // ---- launches (all on stream; graph-capture safe) ----
  { const int n = 2 * 4 * 256;
    zero_kernel<<<(n + 255) / 256, 256, 0, stream>>>(pooled, n); }

  { const size_t tot = (size_t)2 * 4 * 768 * HW;
    pack_x_kernel<<<(unsigned)((tot + 255) / 256), 256, 0, stream>>>(flow, img, res, X); }

  for (int k = 0; k < 6; ++k) {
    const int coC  = (k < 5) ? grow[k] : 256;
    const int coCp = (k < 5) ? 16 : 256;
    size_t tot = (size_t)coCp * 9 * cfg[k].CinPad;
    pack_w_kernel<<<(unsigned)((tot + 255) / 256), 256, 0, stream>>>(
        chW[k], WpCh[k], coC, coCp, cfg[k].Cin, cfg[k].CinPad);
    const int coS = (k < 5) ? grow[k] : 1;
    tot = (size_t)16 * 9 * cfg[k].CinPad;
    pack_w_kernel<<<(unsigned)((tot + 255) / 256), 256, 0, stream>>>(
        spW[k], WpSp[k], coS, 16, cfg[k].Cin, cfg[k].CinPad);
  }

  const size_t Gpred = (size_t)2 * 4 * 28 * HW;   // elements per predictor
  for (int p = 0; p < 2; ++p) {
    bf16_t* Gp = G + p * Gpred;
    for (int k = 0; k < 5; ++k) {
      conv_wmma_kernel<1><<<dim3(384, 1, 2), 192, 0, stream>>>(
          X, Gp, p ? WpSp[k] : WpCh[k], p ? spB[k] : chB[k],
          cfg[k].Cin, cfg[k].CinPad, cfg[k].gIn, cfg[k].gOff,
          /*mode=*/0, grow[k], cfg[k].gOutBase, nullptr, nullptr);
    }
  }
  // final ch conv (Cout=256, 4 N-tiles per block) fused with global average pool
  conv_wmma_kernel<4><<<dim3(384, 4, 2), 192, 0, stream>>>(
      X, G, WpCh[5], chB[5], 796, 800, 28, 0,
      /*mode=*/1, 0, 0, pooled, nullptr);
  // final sp conv (Cout=1) -> spatial map
  conv_wmma_kernel<1><<<dim3(384, 1, 2), 192, 0, stream>>>(
      X, G + Gpred, WpSp[5], spB[5], 796, 800, 28, 0,
      /*mode=*/2, 0, 0, nullptr, sw);

  mlp_kernel<<<2, 256, 0, stream>>>(pooled, chB[5], W1, b1, W2, b2, wgt);
  softmax_kernel<<<8, 256, 0, stream>>>(sw);

  { const size_t tot = (size_t)4 * 256 * HW;
    combine_kernel<<<(unsigned)((tot + 255) / 256), 256, 0, stream>>>(
        flow, res, wgt, sw, (float*)d_out); }
}
